// PyTorchRNNDecoder_9380208575188
// MI455X (gfx1250) — matrix operand order repaired and checked by reading, not compiled, not run on hardware
//
#include <hip/hip_runtime.h>

// ---------------------------------------------------------------------------
// CDNA5 (gfx1250) attention-LSTM decoder. wave32, WMMA bf16 16x16x32 macro-ops.
// ---------------------------------------------------------------------------

typedef __attribute__((ext_vector_type(16))) __bf16         v16bf;
typedef __attribute__((ext_vector_type(8)))  float          v8f;
typedef __attribute__((ext_vector_type(8)))  unsigned short v8u;
typedef __attribute__((ext_vector_type(16))) unsigned short v16u;

#define ENCD 512
#define HID  512
#define ATTD 512
#define VOC  8000
#define NB   32
#define TIL  512
#define TOL  128
#define TSTEPS 129

// ---------------- scalar helpers ----------------
__device__ __forceinline__ unsigned short f2bf(float f) {
  unsigned u = __builtin_bit_cast(unsigned, f);
  u += 0x7FFFu + ((u >> 16) & 1u);             // round-to-nearest-even
  return (unsigned short)(u >> 16);
}
__device__ __forceinline__ float bf2f(unsigned short h) {
  return __builtin_bit_cast(float, ((unsigned)h) << 16);
}
__device__ __forceinline__ float sigmoidf_(float x) {
  return 1.0f / (1.0f + __expf(-x));
}
__device__ __forceinline__ float tanhf_(float x) {
  x = fminf(fmaxf(x, -15.0f), 15.0f);
  float e = __expf(2.0f * x);
  return (e - 1.0f) / (e + 1.0f);
}
__device__ __forceinline__ v8f vzero8() {
  v8f z = {0.f, 0.f, 0.f, 0.f, 0.f, 0.f, 0.f, 0.f};
  return z;
}

// ---------------- WMMA helpers ----------------
__device__ __forceinline__ v8f wmma_bf16(v16bf a, v16bf b, v8f c) {
  // D = A(16x32 bf16) * B(32x16 bf16) + C(16x16 f32)
  return __builtin_amdgcn_wmma_f32_16x16x32_bf16(false, a, false, b,
                                                 (short)0, c, false, false);
}

__device__ __forceinline__ v16bf pack16(v8u lo, v8u hi) {
  v16u t;
#pragma unroll
  for (int i = 0; i < 8; i++) { t[i] = lo[i]; t[i + 8] = hi[i]; }
  return __builtin_bit_cast(v16bf, t);
}

// A-operand (16x32 slab of row-major [rows x ld] bf16 activations).
// ISA layout: lanes 0-15 row=lane, K chunks {0..7},{16..23}; lanes 16-31 +8.
__device__ __forceinline__ v16bf load_A(const unsigned short* base, int ld,
                                        int row0, int kb, int lane) {
  int r  = row0 + (lane & 15);
  int ko = kb + ((lane >> 4) << 3);
  const unsigned short* p = base + (size_t)r * ld + ko;
  return pack16(*(const v8u*)p, *(const v8u*)(p + 16));
}

// B-operand (32x16 slab). B[k][n] = W[n][k] for row-major weight W[out x K].
// Lane = column; lanes 0-15 hold K chunks {0..7},{16..23}; lanes 16-31 +8.
__device__ __forceinline__ v16bf load_B(const unsigned short* base, int ld,
                                        int col0, int kb, int lane) {
  int c  = col0 + (lane & 15);
  int ko = kb + ((lane >> 4) << 3);
  const unsigned short* p = base + (size_t)c * ld + ko;
  return pack16(*(const v8u*)p, *(const v8u*)(p + 16));
}

// ---------------- f32 -> bf16 conversion ----------------
__global__ void f32_to_bf16_kernel(const float* __restrict__ src,
                                   unsigned short* __restrict__ dst, int n) {
  int i = blockIdx.x * blockDim.x + threadIdx.x;
  if (i < n) dst[i] = f2bf(src[i]);
}

// ---------------- keys = enc_pad @ att_k_w^T + b   (16384 x 512, K=512) ----
__global__ void keys_kernel(const unsigned short* __restrict__ A,   // enc bf16
                            const unsigned short* __restrict__ Bw,  // att_k_w bf16
                            const float* __restrict__ bias,
                            float* __restrict__ out) {
  const int lane = threadIdx.x & 31;
  const int wave = ((blockIdx.x * blockDim.x) + threadIdx.x) >> 5;
  const int mt = wave >> 5;   // 1024 row tiles
  const int nt = wave & 31;   // 32 col tiles
  const int m0 = mt * 16, n0 = nt * 16;
  v8f acc = vzero8();
  for (int kb = 0; kb < ENCD; kb += 32) {
    v16bf a = load_A(A, ENCD, m0, kb, lane);
    v16bf b = load_B(Bw, ENCD, n0, kb, lane);
    acc = wmma_bf16(a, b, acc);
  }
  const int nn = n0 + (lane & 15);
  const float bv = bias[nn];
#pragma unroll
  for (int r = 0; r < 8; r++) {
    int m = m0 + r + ((lane >> 4) << 3);
    out[(size_t)m * ATTD + nn] = acc[r] + bv;
  }
}

// ---------------- one LSTM layer, one timestep ----------------
// gates(32 x 2048) = A(32 x (Kih+512)) @ [Wih|Whh]^T + b ; then LSTM pointwise.
// Layer 0 A = [embed[tok] | ctx | h_rec], layers 1/2 A = [inp_low | h_rec].
__global__ void lstm_kernel(int layer, int t,
                            const unsigned short* __restrict__ embbf,
                            const int* __restrict__ tgt,
                            const unsigned short* __restrict__ ctxbf,
                            const unsigned short* __restrict__ h_rec,
                            const unsigned short* __restrict__ inp_low,
                            const unsigned short* __restrict__ Wih, int Kih,
                            const unsigned short* __restrict__ Whh,
                            const float* __restrict__ bih,
                            const float* __restrict__ bhh,
                            float* __restrict__ c_l,
                            unsigned short* __restrict__ h_out) {
  const int lane = threadIdx.x & 31;
  const int wave = ((blockIdx.x * blockDim.x) + threadIdx.x) >> 5;
  const int j0 = wave * 16;                  // 32 waves -> 512 hidden cols
  const int Ktot = Kih + HID;

  auto loadA_step = [&](int row0, int kb) -> v16bf {
    int m  = row0 + (lane & 15);
    int ko = kb + ((lane >> 4) << 3);
    const unsigned short* p;
    if (layer == 0) {
      if (ko < 512) {
        int tok = (t == 0) ? 0 : tgt[m * TOL + (t - 1)];
        p = embbf + (size_t)tok * HID + ko;
      } else if (ko < 1024) {
        p = ctxbf + m * ENCD + (ko - 512);
      } else {
        p = h_rec + m * HID + (ko - 1024);
      }
    } else {
      p = (ko < 512) ? (inp_low + m * HID + ko)
                     : (h_rec + m * HID + (ko - 512));
    }
    return pack16(*(const v8u*)p, *(const v8u*)(p + 16));
  };

  v8f acc[4][2];
#pragma unroll
  for (int g = 0; g < 4; g++) { acc[g][0] = vzero8(); acc[g][1] = vzero8(); }

  for (int kb = 0; kb < Ktot; kb += 32) {
    v16bf a0 = loadA_step(0, kb);
    v16bf a1 = loadA_step(16, kb);
    const unsigned short* W;
    int ld, kk;
    if (kb < Kih) { W = Wih; ld = Kih; kk = kb; }
    else          { W = Whh; ld = HID; kk = kb - Kih; }
#pragma unroll
    for (int g = 0; g < 4; g++) {
      v16bf b = load_B(W, ld, g * HID + j0, kk, lane);
      acc[g][0] = wmma_bf16(a0, b, acc[g][0]);
      acc[g][1] = wmma_bf16(a1, b, acc[g][1]);
    }
  }

  const int ncol = j0 + (lane & 15);
  float bsum[4];
#pragma unroll
  for (int g = 0; g < 4; g++) bsum[g] = bih[g * HID + ncol] + bhh[g * HID + ncol];

#pragma unroll
  for (int mt = 0; mt < 2; mt++) {
#pragma unroll
    for (int r = 0; r < 8; r++) {
      int m = mt * 16 + r + ((lane >> 4) << 3);
      float gi = acc[0][mt][r] + bsum[0];
      float gf = acc[1][mt][r] + bsum[1];
      float gg = acc[2][mt][r] + bsum[2];
      float go = acc[3][mt][r] + bsum[3];
      size_t idx = (size_t)m * HID + ncol;
      float co = c_l[idx];
      float cn = sigmoidf_(gf) * co + sigmoidf_(gi) * tanhf_(gg);
      c_l[idx] = cn;
      h_out[idx] = f2bf(sigmoidf_(go) * tanhf_(cn));
    }
  }
}

// ---------------- attention + softmax + context + proj (per batch row) -----
__global__ void att_kernel(const unsigned short* __restrict__ h2bf,
                           const float* __restrict__ qw,
                           const float* __restrict__ qb,
                           const float* __restrict__ keys,
                           const int* __restrict__ enc_len,
                           const float* __restrict__ enc_pad,
                           const float* __restrict__ projw,
                           const float* __restrict__ projb,
                           unsigned short* __restrict__ ctxbf,
                           unsigned short* __restrict__ psbf,
                           float* __restrict__ alis, int t) {
  __shared__ float sh_h[HID];
  __shared__ float sh_q[ATTD];
  __shared__ float sh_ali[TIL];
  __shared__ float sh_ctx[ENCD];
  __shared__ float red1[16];
  __shared__ float red2[16];

  const int n = blockIdx.x;
  const int tid = threadIdx.x;   // 512 threads

  sh_h[tid] = bf2f(h2bf[n * HID + tid]);
  __syncthreads();

  // q = att_q_w @ h + b
  {
    float a = qb[tid];
    const float* wr = qw + (size_t)tid * HID;
    for (int k = 0; k < HID; k++) a += wr[k] * sh_h[k];
    sh_q[tid] = a;
  }
  __syncthreads();

  // e[t] = keys[n,t,:] . q * scale, masked
  float e;
  {
    const float* kr = keys + ((size_t)n * TIL + tid) * ATTD;
    float a = 0.f;
    for (int k = 0; k < ATTD; k++) a += kr[k] * sh_q[k];
    e = a * 0.044194173824159216f;  // 1/sqrt(512)
    if (tid >= enc_len[n]) e = -1e30f;
  }

  // block max (wave32 shuffles + LDS)
  float m = e;
#pragma unroll
  for (int off = 16; off >= 1; off >>= 1) m = fmaxf(m, __shfl_xor(m, off, 32));
  if ((tid & 31) == 0) red1[tid >> 5] = m;
  __syncthreads();
  float mx = -1e30f;
  for (int i = 0; i < 16; i++) mx = fmaxf(mx, red1[i]);

  float ex = __expf(e - mx);
  float s = ex;
#pragma unroll
  for (int off = 16; off >= 1; off >>= 1) s += __shfl_xor(s, off, 32);
  if ((tid & 31) == 0) red2[tid >> 5] = s;
  __syncthreads();
  float sum = 0.f;
  for (int i = 0; i < 16; i++) sum += red2[i];

  float ali = ex / sum;
  sh_ali[tid] = ali;
  alis[((size_t)n * TSTEPS + t) * TIL + tid] = ali;
  __syncthreads();

  // ctx[d] = sum_t ali[t] * enc_pad[n,t,d]
  {
    float cx = 0.f;
    const float* ep = enc_pad + (size_t)n * TIL * ENCD + tid;
    for (int tt = 0; tt < TIL; tt++) cx += sh_ali[tt] * ep[(size_t)tt * ENCD];
    sh_ctx[tid] = cx;
    ctxbf[n * ENCD + tid] = f2bf(cx);
  }
  __syncthreads();

  // p = relu(proj_w @ [h|ctx] + b) -> bf16 state for vocab GEMM
  {
    const float* pr = projw + (size_t)tid * (HID + ENCD);
    float p = projb[tid];
    for (int k = 0; k < HID; k++)  p += pr[k] * sh_h[k];
    for (int k = 0; k < ENCD; k++) p += pr[HID + k] * sh_ctx[k];
    p = fmaxf(p, 0.f);
    psbf[n * HID + tid] = f2bf(p);
  }
}

// ---------------- pred = pstate(32x512) @ pred_w^T (8000) + b --------------
__global__ void pred_kernel(const unsigned short* __restrict__ ps,
                            const unsigned short* __restrict__ Wv,
                            const float* __restrict__ bias,
                            float* __restrict__ outs, int t) {
  const int lane = threadIdx.x & 31;
  const int wave = ((blockIdx.x * blockDim.x) + threadIdx.x) >> 5;
  if (wave >= VOC / 16) return;            // wave-uniform: EXEC stays all-ones
  const int n0 = wave * 16;
  v8f acc0 = vzero8(), acc1 = vzero8();
  for (int kb = 0; kb < HID; kb += 32) {
    v16bf a0 = load_A(ps, HID, 0, kb, lane);
    v16bf a1 = load_A(ps, HID, 16, kb, lane);
    v16bf b  = load_B(Wv, HID, n0, kb, lane);
    acc0 = wmma_bf16(a0, b, acc0);
    acc1 = wmma_bf16(a1, b, acc1);
  }
  const int v = n0 + (lane & 15);
  const float bv = bias[v];
#pragma unroll
  for (int r = 0; r < 8; r++) {
    int m0 = r + ((lane >> 4) << 3);
    outs[((size_t)m0 * TSTEPS + t) * VOC + v]        = acc0[r] + bv;
    outs[((size_t)(m0 + 16) * TSTEPS + t) * VOC + v] = acc1[r] + bv;
  }
}

// ---------------------------------------------------------------------------
extern "C" void kernel_launch(void* const* d_in, const int* in_sizes, int n_in,
                              void* d_out, int out_size, void* d_ws,
                              size_t ws_size, hipStream_t stream) {
  (void)in_sizes; (void)n_in; (void)out_size; (void)ws_size;

  const float* enc_pad = (const float*)d_in[0];
  const int*   enc_len = (const int*)d_in[1];
  const int*   tgt     = (const int*)d_in[2];
  const float* embed   = (const float*)d_in[3];
  const float* Wih[3]  = {(const float*)d_in[4], (const float*)d_in[8],  (const float*)d_in[12]};
  const float* Whh[3]  = {(const float*)d_in[5], (const float*)d_in[9],  (const float*)d_in[13]};
  const float* bih[3]  = {(const float*)d_in[6], (const float*)d_in[10], (const float*)d_in[14]};
  const float* bhh[3]  = {(const float*)d_in[7], (const float*)d_in[11], (const float*)d_in[15]};
  const float* att_k_w = (const float*)d_in[16];
  const float* att_k_b = (const float*)d_in[17];
  const float* att_q_w = (const float*)d_in[18];
  const float* att_q_b = (const float*)d_in[19];
  const float* proj_w  = (const float*)d_in[20];
  const float* proj_b  = (const float*)d_in[21];
  const float* pred_w  = (const float*)d_in[22];
  const float* pred_b  = (const float*)d_in[23];

  // ---- workspace carve (all 256B-aligned) ----
  char* base = (char*)d_ws;
  size_t off = 0;
  auto carve = [&](size_t elems, size_t esz) -> void* {
    void* p = base + off;
    off += (elems * esz + 255) & ~(size_t)255;
    return p;
  };
  unsigned short* emb_bf   = (unsigned short*)carve((size_t)VOC * HID, 2);
  unsigned short* enc_bf   = (unsigned short*)carve((size_t)NB * TIL * ENCD, 2);
  unsigned short* wih_bf0  = (unsigned short*)carve(2048 * 1024, 2);
  unsigned short* wih_bf1  = (unsigned short*)carve(2048 * 512, 2);
  unsigned short* wih_bf2  = (unsigned short*)carve(2048 * 512, 2);
  unsigned short* whh_bf0  = (unsigned short*)carve(2048 * 512, 2);
  unsigned short* whh_bf1  = (unsigned short*)carve(2048 * 512, 2);
  unsigned short* whh_bf2  = (unsigned short*)carve(2048 * 512, 2);
  unsigned short* attk_bf  = (unsigned short*)carve(ATTD * ENCD, 2);
  unsigned short* predw_bf = (unsigned short*)carve((size_t)VOC * ENCD, 2);
  float*          keys     = (float*)carve((size_t)NB * TIL * ATTD, 4);
  float*          c_all    = (float*)carve(3 * NB * HID, 4);
  unsigned short* h_bf     = (unsigned short*)carve(2 * 3 * NB * HID, 2);
  unsigned short* ctx_bf   = (unsigned short*)carve(NB * ENCD, 2);
  unsigned short* ps_bf    = (unsigned short*)carve(NB * HID, 2);

  auto conv = [&](const float* s, unsigned short* d, int n) {
    f32_to_bf16_kernel<<<(n + 255) / 256, 256, 0, stream>>>(s, d, n);
  };
  conv(embed,    emb_bf,   VOC * HID);
  conv(enc_pad,  enc_bf,   NB * TIL * ENCD);
  conv(Wih[0],   wih_bf0,  2048 * 1024);
  conv(Wih[1],   wih_bf1,  2048 * 512);
  conv(Wih[2],   wih_bf2,  2048 * 512);
  conv(Whh[0],   whh_bf0,  2048 * 512);
  conv(Whh[1],   whh_bf1,  2048 * 512);
  conv(Whh[2],   whh_bf2,  2048 * 512);
  conv(att_k_w,  attk_bf,  ATTD * ENCD);
  conv(pred_w,   predw_bf, VOC * ENCD);

  hipMemsetAsync(c_all,  0, (size_t)3 * NB * HID * 4, stream);
  hipMemsetAsync(h_bf,   0, (size_t)2 * 3 * NB * HID * 2, stream);
  hipMemsetAsync(ctx_bf, 0, (size_t)NB * ENCD * 2, stream);

  // precompute attention keys: 16384x512 @ 512x512 (bf16 WMMA)
  keys_kernel<<<4096, 256, 0, stream>>>(enc_bf, attk_bf, att_k_b, keys);

  float* outs = (float*)d_out;
  float* alis = outs + (size_t)NB * TSTEPS * VOC;

  const unsigned short* wihb[3] = {wih_bf0, wih_bf1, wih_bf2};
  const unsigned short* whhb[3] = {whh_bf0, whh_bf1, whh_bf2};
  const int kih[3] = {1024, 512, 512};
  const size_t HS = (size_t)NB * HID;

  for (int t = 0; t < TSTEPS; ++t) {
    unsigned short* hR = h_bf + (size_t)(t & 1) * 3 * HS;        // prev step
    unsigned short* hW = h_bf + (size_t)((t & 1) ^ 1) * 3 * HS;  // this step
    for (int l = 0; l < 3; ++l) {
      lstm_kernel<<<4, 256, 0, stream>>>(
          l, t, emb_bf, tgt, ctx_bf,
          hR + (size_t)l * HS,
          (l == 0) ? (const unsigned short*)nullptr : hW + (size_t)(l - 1) * HS,
          wihb[l], kih[l], whhb[l], bih[l], bhh[l],
          c_all + (size_t)l * HS, hW + (size_t)l * HS);
    }
    att_kernel<<<NB, 512, 0, stream>>>(hW + 2 * HS, att_q_w, att_q_b, keys,
                                       enc_len, enc_pad, proj_w, proj_b,
                                       ctx_bf, ps_bf, alis, t);
    pred_kernel<<<63, 256, 0, stream>>>(ps_bf, predw_bf, pred_b, outs, t);
  }
}
